// AttentionAggregator_35502199669476
// MI455X (gfx1250) — matrix operand order repaired and checked by reading, not compiled, not run on hardware
//
#include <hip/hip_runtime.h>

#define B_  16
#define N_  2048
#define DIN 128
#define DQ  64

typedef __attribute__((ext_vector_type(16))) _Float16 v16h;
typedef __attribute__((ext_vector_type(8)))  _Float16 v8h;
typedef __attribute__((ext_vector_type(8)))  float    v8f;

#define LOG2E 1.44269504088896340736f

union V16 { v16h v; v8h h[2]; };

// Butterfly max over each 16-lane group using DPP (pure VALU, no LDS):
// xor1 = quad_perm(1,0,3,2), xor2 = quad_perm(2,3,0,1),
// xor7 = row_half_mirror (0x141), xor15 = row_mirror (0x140).
__device__ __forceinline__ float dpp_max16(float v) {
  float m = v, t;
  t = __builtin_bit_cast(float, __builtin_amdgcn_update_dpp(
          0, __builtin_bit_cast(int, m), 0xB1, 0xF, 0xF, true));
  m = fmaxf(m, t);
  t = __builtin_bit_cast(float, __builtin_amdgcn_update_dpp(
          0, __builtin_bit_cast(int, m), 0x4E, 0xF, 0xF, true));
  m = fmaxf(m, t);
  t = __builtin_bit_cast(float, __builtin_amdgcn_update_dpp(
          0, __builtin_bit_cast(int, m), 0x141, 0xF, 0xF, true));
  m = fmaxf(m, t);
  t = __builtin_bit_cast(float, __builtin_amdgcn_update_dpp(
          0, __builtin_bit_cast(int, m), 0x140, 0xF, 0xF, true));
  m = fmaxf(m, t);
  return m;
}

// ---------------------------------------------------------------------------
// Kernel 1: Q/K projections. K is pre-scaled by log2(e)/sqrt(DQ) so the whole
// softmax runs in base 2 (v_exp_f32 IS exp2 on this HW -> no pre-multiply).
// ---------------------------------------------------------------------------
__global__ __launch_bounds__(256) void qk_proj_kernel(
    const float* __restrict__ x,
    const float* __restrict__ Wq, const float* __restrict__ bq,
    const float* __restrict__ Wk, const float* __restrict__ bk,
    _Float16* __restrict__ Qh, _Float16* __restrict__ Kh)
{
  int id = blockIdx.x * 256 + threadIdx.x;   // B*N*DQ threads
  int q  = id & (DQ - 1);
  int ng = id >> 6;                          // flat (b*N + n)
  const float4* xr = (const float4*)(x  + (size_t)ng * DIN);
  const float4* wq = (const float4*)(Wq + (size_t)q  * DIN);
  const float4* wk = (const float4*)(Wk + (size_t)q  * DIN);
  float aq = 0.f, ak = 0.f;
#pragma unroll
  for (int i = 0; i < DIN / 4; ++i) {
    float4 xv = xr[i], a = wq[i], b = wk[i];
    aq += xv.x * a.x + xv.y * a.y + xv.z * a.z + xv.w * a.w;
    ak += xv.x * b.x + xv.y * b.y + xv.z * b.z + xv.w * b.w;
  }
  aq += bq[q];
  ak  = (ak + bk[q]) * (0.125f * LOG2E);
  Qh[(size_t)ng * DQ + q] = (_Float16)aq;
  Kh[(size_t)ng * DQ + q] = (_Float16)ak;
}

// ---------------------------------------------------------------------------
// Kernel 2: fused masked-softmax attention, flash style, double-buffered LDS.
// 256 threads = 8 waves; wave w owns 16 output rows; all waves share the
// staged Q / X^T tiles. Softmax denominator accumulated by a 9th WMMA with a
// constant ones B-fragment (no cross-lane sum reductions needed).
// ---------------------------------------------------------------------------
#define QSTR 72   // 64 q halves + 8 pad (16B-aligned rows, conflict-free b128)
#define XSTR 40   // 32 m halves + 8 pad
#define PSTR 40

__global__ __launch_bounds__(256) void attn_kernel(
    const float* __restrict__ x, const int* __restrict__ mask,
    const _Float16* __restrict__ Qh, const _Float16* __restrict__ Kh,
    float* __restrict__ out)
{
  __shared__ __align__(16) _Float16 lds_q[2][32 * QSTR];    // Q tile [m=32][q]
  __shared__ __align__(16) _Float16 lds_xt[2][DIN * XSTR];  // X^T tile [d][m=32]
  __shared__ __align__(16) _Float16 lds_p[8][16 * PSTR];    // per-wave P bounce
  __shared__ float lds_neg[2][32];

  const int t    = threadIdx.x;
  const int wave = t >> 5;
  const int lane = t & 31;
  const int b    = blockIdx.x >> 4;          // 16 workgroups per batch
  const int tile = blockIdx.x & 15;
  const int row0 = tile * 128 + wave * 16;

  const int col  = lane & 15;
  const int hi   = (lane >> 4) & 1;
  const int aoff = hi ? 8  : 0;              // unused; A-frags also use boff
  const int boff = hi ? 16 : 0;              // B-frag k sub-offset

  // staging assignments
  const int sq_r  = t >> 3;                  // Q tile row 0..31
  const int sq_c  = (t & 7) * 8;             // Q tile col group
  const int sx_d0 = (t & 15) * 8;            // X d group (coalesced globals)
  const int sx_m0 = (t >> 4) * 2;            // X m pair (packed b32 LDS stores)

  const size_t xbase = (size_t)b * N_ * DIN;
  const size_t qbase = (size_t)b * N_ * DQ;

  // K A-fragments for this wave's rows (resident across the whole loop)
  V16 aK0, aK1;
  {
    const _Float16* kr = Kh + qbase + (size_t)(row0 + col) * DQ;
    aK0.h[0] = *(const v8h*)(kr + 0  + boff);
    aK0.h[1] = *(const v8h*)(kr + 8  + boff);
    aK1.h[0] = *(const v8h*)(kr + 32 + boff);
    aK1.h[1] = *(const v8h*)(kr + 40 + boff);
  }

  // ones B-fragment: O[8] accumulates the softmax denominator (P . 1)
  V16 onesf;
#pragma unroll
  for (int i = 0; i < 16; ++i) onesf.v[i] = (_Float16)1.0f;

  v8f O[9] = {};
  float rmax[8];
#pragma unroll
  for (int i = 0; i < 8; ++i) rmax[i] = -1e30f;

  // ---- prologue: stage chunk 0 into buffer 0 ----
  {
    *(v8h*)&lds_q[0][sq_r * QSTR + sq_c] =
        *(const v8h*)(Qh + qbase + (size_t)sq_r * DQ + sq_c);
    const float4* r0 = (const float4*)(x + xbase + (size_t)sx_m0 * DIN + sx_d0);
    const float4* r1 = (const float4*)(x + xbase + (size_t)(sx_m0 + 1) * DIN + sx_d0);
    float4 a0 = r0[0], a1 = r0[1], c0v = r1[0], c1v = r1[1];
    float fa[8] = {a0.x, a0.y, a0.z, a0.w, a1.x, a1.y, a1.z, a1.w};
    float fb[8] = {c0v.x, c0v.y, c0v.z, c0v.w, c1v.x, c1v.y, c1v.z, c1v.w};
#pragma unroll
    for (int e = 0; e < 8; ++e) {
      union { _Float16 h[2]; unsigned u; } pk;
      pk.h[0] = (_Float16)fa[e];
      pk.h[1] = (_Float16)fb[e];
      *(unsigned*)&lds_xt[0][(sx_d0 + e) * XSTR + sx_m0] = pk.u;
    }
    // base-2 mask bias: -1e9 * log2(e)
    if (t < 32) lds_neg[0][t] = (mask[(size_t)b * N_ + t] > 0) ? 0.f : -1.442695e9f;
  }
  __syncthreads();

  for (int i = 0; i < N_ / 32; ++i) {
    const int cur = i & 1, nxt = cur ^ 1;
    const int c1  = (i + 1) * 32;
    const bool have_next = (c1 < N_);

    // ---- issue next chunk's global loads early (latency hidden by WMMAs) --
    v8h qstage = {};
    float fa[8], fb[8];
    float nstage = 0.f;
    if (have_next) {
      qstage = *(const v8h*)(Qh + qbase + (size_t)(c1 + sq_r) * DQ + sq_c);
      const float4* r0 = (const float4*)(x + xbase + (size_t)(c1 + sx_m0) * DIN + sx_d0);
      const float4* r1 = (const float4*)(x + xbase + (size_t)(c1 + sx_m0 + 1) * DIN + sx_d0);
      float4 a0 = r0[0], a1 = r0[1], c0v = r1[0], c1v = r1[1];
      fa[0] = a0.x; fa[1] = a0.y; fa[2] = a0.z; fa[3] = a0.w;
      fa[4] = a1.x; fa[5] = a1.y; fa[6] = a1.z; fa[7] = a1.w;
      fb[0] = c0v.x; fb[1] = c0v.y; fb[2] = c0v.z; fb[3] = c0v.w;
      fb[4] = c1v.x; fb[5] = c1v.y; fb[6] = c1v.z; fb[7] = c1v.w;
      if (t < 32) nstage = (mask[(size_t)b * N_ + c1 + t] > 0) ? 0.f : -1.442695e9f;
      if (c1 + 32 < N_) {
        __builtin_prefetch(x  + xbase + (size_t)(c1 + 32 + (t >> 4)) * DIN, 0, 3);
        __builtin_prefetch(Qh + qbase + (size_t)(c1 + 32 + (t >> 3)) * DQ, 0, 3);
      }
    }

    // ---------------- S = K . Q^T  (16 x 32 chunk, base-2 scaled) ---------
    const _Float16* lq = lds_q[cur];
    const _Float16* lx = lds_xt[cur];
    V16 b00, b01, b10, b11;
    {
      const _Float16* q0 = lq + col * QSTR;
      const _Float16* q1 = lq + (16 + col) * QSTR;
      b00.h[0] = *(const v8h*)(q0 + 0  + boff);  b00.h[1] = *(const v8h*)(q0 + 8  + boff);
      b01.h[0] = *(const v8h*)(q0 + 32 + boff);  b01.h[1] = *(const v8h*)(q0 + 40 + boff);
      b10.h[0] = *(const v8h*)(q1 + 0  + boff);  b10.h[1] = *(const v8h*)(q1 + 8  + boff);
      b11.h[0] = *(const v8h*)(q1 + 32 + boff);  b11.h[1] = *(const v8h*)(q1 + 40 + boff);
    }
    v8f s0 = {}, s1 = {};
    s0 = __builtin_amdgcn_wmma_f32_16x16x32_f16(false, aK0.v, false, b00.v, (short)0, s0, false, false);
    s0 = __builtin_amdgcn_wmma_f32_16x16x32_f16(false, aK1.v, false, b01.v, (short)0, s0, false, false);
    s1 = __builtin_amdgcn_wmma_f32_16x16x32_f16(false, aK0.v, false, b10.v, (short)0, s1, false, false);
    s1 = __builtin_amdgcn_wmma_f32_16x16x32_f16(false, aK1.v, false, b11.v, (short)0, s1, false, false);

    const float neg0 = lds_neg[cur][col];
    const float neg1 = lds_neg[cur][16 + col];

    // ------- online softmax in base 2 (per-row = per-VGPR element) --------
    float al[8];
    _Float16 p0h[8], p1h[8];
    bool resc = false;
#pragma unroll
    for (int r = 0; r < 8; ++r) {
      float v0 = s0[r] + neg0;     // scores already carry log2e/sqrt(DQ)
      float v1 = s1[r] + neg1;
      float cm = dpp_max16(fmaxf(v0, v1));
      float nm = fmaxf(rmax[r], cm);
      float a  = __builtin_amdgcn_exp2f(rmax[r] - nm);   // v_exp_f32
      rmax[r]  = nm;
      p0h[r] = (_Float16)__builtin_amdgcn_exp2f(v0 - nm);
      p1h[r] = (_Float16)__builtin_amdgcn_exp2f(v1 - nm);
      al[r]  = a;
      resc  |= (a < 1.0f);
    }
    if (__any(resc)) {             // wave-uniform: skip rescale when max stable
#pragma unroll
      for (int blk = 0; blk < 9; ++blk)
#pragma unroll
        for (int r = 0; r < 8; ++r)
          O[blk][r] *= al[r];
    }

    // P: C-layout -> A-layout via per-wave LDS bounce
    {
      _Float16* pw = &lds_p[wave][0];
      int prow = hi * 8;
#pragma unroll
      for (int r = 0; r < 8; ++r) {
        pw[(prow + r) * PSTR + col]      = p0h[r];
        pw[(prow + r) * PSTR + 16 + col] = p1h[r];
      }
    }
    asm volatile("s_wait_dscnt 0" ::: "memory");  // DS in-order within wave

    V16 ap;
    {
      const _Float16* pr = &lds_p[wave][col * PSTR];
      ap.h[0] = *(const v8h*)(pr + boff);
      ap.h[1] = *(const v8h*)(pr + 8 + boff);
    }

    // ---- denominator: O[8] += P . ones ----
    O[8] = __builtin_amdgcn_wmma_f32_16x16x32_f16(false, ap.v, false, onesf.v, (short)0, O[8], false, false);

    // ---- O += P . X (8 column blocks of 16) ----
#pragma unroll
    for (int blk = 0; blk < 8; ++blk) {
      V16 bx;
      const _Float16* xr = lx + (blk * 16 + col) * XSTR + boff;
      bx.h[0] = *(const v8h*)(xr);
      bx.h[1] = *(const v8h*)(xr + 8);
      O[blk] = __builtin_amdgcn_wmma_f32_16x16x32_f16(false, ap.v, false, bx.v, (short)0, O[blk], false, false);
    }

    // ---- store next chunk into the other buffer (no extra barrier) ----
    if (have_next) {
      *(v8h*)&lds_q[nxt][sq_r * QSTR + sq_c] = qstage;
#pragma unroll
      for (int e = 0; e < 8; ++e) {
        union { _Float16 h[2]; unsigned u; } pk;
        pk.h[0] = (_Float16)fa[e];
        pk.h[1] = (_Float16)fb[e];
        *(unsigned*)&lds_xt[nxt][(sx_d0 + e) * XSTR + sx_m0] = pk.u;
      }
      if (t < 32) lds_neg[nxt][t] = nstage;
    }
    __syncthreads();
  }

  // ---------------- epilogue: out = O / denom ----------------
#pragma unroll
  for (int r = 0; r < 8; ++r) {
    float inv = 1.0f / O[8][r];
    size_t ro = ((size_t)b * N_ + row0 + hi * 8 + r) * DIN;
#pragma unroll
    for (int blk = 0; blk < 8; ++blk)
      out[ro + blk * 16 + col] = O[blk][r] * inv;
  }
}

// ---------------------------------------------------------------------------
extern "C" void kernel_launch(void* const* d_in, const int* in_sizes, int n_in,
                              void* d_out, int out_size, void* d_ws, size_t ws_size,
                              hipStream_t stream)
{
  (void)in_sizes; (void)n_in; (void)out_size; (void)ws_size;
  const float* x    = (const float*)d_in[0];
  const int*   mask = (const int*)d_in[1];
  const float* Wq   = (const float*)d_in[2];
  const float* bq   = (const float*)d_in[3];
  const float* Wk   = (const float*)d_in[4];
  const float* bk   = (const float*)d_in[5];
  float* out = (float*)d_out;

  // f16 Q/K workspace: 2 * B*N*DQ * 2B = 8 MB
  _Float16* Qh = (_Float16*)d_ws;
  _Float16* Kh = Qh + (size_t)B_ * N_ * DQ;

  qk_proj_kernel<<<(B_ * N_ * DQ) / 256, 256, 0, stream>>>(x, Wq, bq, Wk, bk, Qh, Kh);
  attn_kernel<<<B_ * 16, 256, 0, stream>>>(x, mask, Qh, Kh, out);
}
